// FlashQwenAttention_8718783611708
// MI455X (gfx1250) — matrix operand order repaired and checked by reading, not compiled, not run on hardware
//
#include <hip/hip_runtime.h>
#include <hip/hip_bf16.h>

// Problem constants (match reference)
#define T_TOK 4096
#define SEQ   1024
#define NB    4
#define NH    16
#define HD    128
#define HID   2048
#define HID3  6144
#define HALF_D 64

typedef __attribute__((ext_vector_type(16))) __bf16 v16bf;
typedef __attribute__((ext_vector_type(8)))  float  v8f;

static __device__ __forceinline__ v8f vzero8() {
  v8f z;
#pragma unroll
  for (int i = 0; i < 8; ++i) z[i] = 0.0f;
  return z;
}

static __device__ __forceinline__ v8f wmma_bf16(v16bf a, v16bf b, v8f c) {
  // (neg_a, A, neg_b, B, c_mod, C, reuse_a, reuse_b)
  return __builtin_amdgcn_wmma_f32_16x16x32_bf16(false, a, false, b, (short)0, c,
                                                 false, false);
}

// A fragment: 16x32 bf16 (MxK), source row-major with leading dim ld.
// Layout: lanes 0-15 row M=lane, K = 0..15; lanes 16-31 same rows,
// K = 16..31 (contiguous, matching the B fragments).
static __device__ __forceinline__ v16bf load_a16(const __bf16* base, int ld) {
  const int lane = threadIdx.x & 31;
  const int r  = lane & 15;
  const int hi = lane >> 4;
  const __bf16* p = base + (size_t)r * ld + hi * 16;
  v16bf a;
#pragma unroll
  for (int i = 0; i < 8; ++i) a[i] = p[i];
#pragma unroll
  for (int i = 0; i < 8; ++i) a[8 + i] = p[8 + i];
  return a;
}

// B fragment: 32x16 bf16 (KxN) with B[k][n] = src[n*ld + k]  (src is N x K
// row-major, e.g. a weight matrix accessed as W^T).  Layout: lane n
// (=lane&15), halves hold K = 0..15 / 16..31, contiguous.
static __device__ __forceinline__ v16bf load_bT16(const __bf16* base, int ld) {
  const int lane = threadIdx.x & 31;
  const int n  = lane & 15;
  const int hi = lane >> 4;
  const __bf16* p = base + (size_t)n * ld + hi * 16;
  v16bf b;
#pragma unroll
  for (int i = 0; i < 16; ++i) b[i] = p[i];
  return b;
}

// ---------------------------------------------------------------------------
__global__ void f32_to_bf16_kernel(const float* __restrict__ in,
                                   __bf16* __restrict__ out, int n) {
  int i = blockIdx.x * blockDim.x + threadIdx.x;
  if (i < n) out[i] = (__bf16)in[i];
}

// ---------------------------------------------------------------------------
// QKV GEMM + bias + RoPE.  One wave computes a 32(token) x 128(head-dim) tile
// (two 16-row groups sharing every B fragment -> 2x arithmetic intensity on W).
// q,k stored [B,H,S,D]; v stored transposed [B,H,D,S] so attention's PV
// A-fragments are contiguous.
__global__ void qkv_rope_kernel(const __bf16* __restrict__ A,   // T x HID bf16
                                const __bf16* __restrict__ W,   // 3HID x HID bf16
                                const float* __restrict__ bias, // 3HID
                                const float* __restrict__ cosb, // T x 64
                                const float* __restrict__ sinb, // T x 64
                                __bf16* __restrict__ qo,
                                __bf16* __restrict__ ko,
                                __bf16* __restrict__ vTo) {
  const int wave = (blockIdx.x * blockDim.x + threadIdx.x) >> 5;
  const int rt  = wave / 48;        // token tile (32 tokens)
  const int ch  = wave % 48;        // 0..47: [q heads | k heads | v heads]
  const int typ = ch >> 4;          // 0=q 1=k 2=v
  const int h   = ch & 15;
  const int ncol0 = ch * HD;
  const int lane = threadIdx.x & 31;
  const int n = lane & 15, hiL = lane >> 4;

  v8f acc[2][8];
#pragma unroll
  for (int g = 0; g < 2; ++g)
#pragma unroll
    for (int i = 0; i < 8; ++i) acc[g][i] = vzero8();

  const __bf16* Arow0 = A + (size_t)rt * 32 * HID;
  const __bf16* Arow1 = Arow0 + (size_t)16 * HID;
  for (int kk = 0; kk < HID; kk += 32) {
    v16bf a0 = load_a16(Arow0 + kk, HID);
    v16bf a1 = load_a16(Arow1 + kk, HID);
#pragma unroll
    for (int nt = 0; nt < 8; ++nt) {
      v16bf b = load_bT16(W + (size_t)(ncol0 + nt * 16) * HID + kk, HID);
      acc[0][nt] = wmma_bf16(a0, b, acc[0][nt]);
      acc[1][nt] = wmma_bf16(a1, b, acc[1][nt]);
    }
  }

  if (typ < 2) {
    // q/k: bias + RoPE (pair d with d+64 -> frag nt with nt+4, same lane)
    const float scale = (typ == 0) ? 0.08838834764831845f : 1.0f; // 128^-0.5 on q
    __bf16* dst = (typ == 0) ? qo : ko;
#pragma unroll
    for (int g = 0; g < 2; ++g) {
#pragma unroll
      for (int nt = 0; nt < 4; ++nt) {
        const int dloc = nt * 16 + n;              // 0..63
#pragma unroll
        for (int i = 0; i < 8; ++i) {
          const int t  = rt * 32 + g * 16 + i + hiL * 8;   // global token
          const int b0 = t / SEQ;
          const int sidx = t - b0 * SEQ;
          const float c = cosb[(size_t)t * HALF_D + dloc];
          const float s = sinb[(size_t)t * HALF_D + dloc];
          const float x1 = acc[g][nt][i]     + bias[ncol0 + dloc];
          const float x2 = acc[g][nt + 4][i] + bias[ncol0 + dloc + 64];
          const float r1 = (x1 * c - x2 * s) * scale;
          const float r2 = (x2 * c + x1 * s) * scale;
          const size_t base = (((size_t)(b0 * NH + h) * SEQ) + sidx) * HD;
          dst[base + dloc]      = (__bf16)r1;
          dst[base + dloc + 64] = (__bf16)r2;
        }
      }
    }
  } else {
    // v: bias only, store transposed [B,H,D,S]
#pragma unroll
    for (int g = 0; g < 2; ++g) {
#pragma unroll
      for (int nt = 0; nt < 8; ++nt) {
        const int dloc = nt * 16 + n;              // 0..127
#pragma unroll
        for (int i = 0; i < 8; ++i) {
          const int t  = rt * 32 + g * 16 + i + hiL * 8;
          const int b0 = t / SEQ;
          const int sidx = t - b0 * SEQ;
          const float v = acc[g][nt][i] + bias[ncol0 + dloc];
          vTo[(((size_t)(b0 * NH + h) * HD) + dloc) * SEQ + sidx] = (__bf16)v;
        }
      }
    }
  }
}

// ---------------------------------------------------------------------------
// Causal flash attention.  One wave owns 16 queries of one (b,h).
// S^T = K * Q^T so per-query softmax stats are in-lane (query = lane&15,
// shared with lane^16).  PV uses V^T (preloaded transposed) as A and P^T as B.
__global__ void attn_kernel(const __bf16* __restrict__ Q,   // [B,H,S,D]
                            const __bf16* __restrict__ K,   // [B,H,S,D]
                            const __bf16* __restrict__ VT,  // [B,H,D,S]
                            __bf16* __restrict__ O) {       // [T,HID] bf16
  const int bh   = blockIdx.x >> 3;
  const int qblk = blockIdx.x & 7;
  const int wave = threadIdx.x >> 5;
  const int q0   = qblk * 128 + wave * 16;
  const int lane = threadIdx.x & 31;
  const int n = lane & 15, hiL = lane >> 4;

  const __bf16* Qh = Q  + (size_t)bh * SEQ * HD;
  const __bf16* Kh = K  + (size_t)bh * SEQ * HD;
  const __bf16* Vh = VT + (size_t)bh * HD * SEQ;

  v16bf qf[4];
#pragma unroll
  for (int d = 0; d < 4; ++d)
    qf[d] = load_bT16(Qh + (size_t)q0 * HD + d * 32, HD);

  v8f acc[8];                    // O^T: 8 d-tiles x (16 dim x 16 query)
#pragma unroll
  for (int i = 0; i < 8; ++i) acc[i] = vzero8();

  float m = -1e30f, l = 0.0f;
  const int q = q0 + n;          // this lane's query index (within seq)

  for (int k0 = 0; k0 <= q0 + 15; k0 += 32) {
    // Prefetch next chunk of K (8KB) and part of V^T into cache
    // (global_prefetch_b8; lane-strided so a wave covers many lines).
    if (k0 + 32 <= q0 + 15) {
      __builtin_prefetch(Kh + (size_t)(k0 + 32) * HD + (lane << 7), 0, 0);
      __builtin_prefetch(Vh + (size_t)(lane << 2) * SEQ + (k0 + 32), 0, 0);
    }

    // ---- S^T for 32 keys: two 16x16 tiles ----
    v8f s0 = vzero8(), s1 = vzero8();
#pragma unroll
    for (int d = 0; d < 4; ++d) {
      v16bf ka = load_a16(Kh + (size_t)k0 * HD + d * 32, HD);
      s0 = wmma_bf16(ka, qf[d], s0);
    }
#pragma unroll
    for (int d = 0; d < 4; ++d) {
      v16bf ka = load_a16(Kh + (size_t)(k0 + 16) * HD + d * 32, HD);
      s1 = wmma_bf16(ka, qf[d], s1);
    }
    float p0[8], p1[8];
#pragma unroll
    for (int i = 0; i < 8; ++i) { p0[i] = s0[i]; p1[i] = s1[i]; }

    // ---- causal mask (diagonal chunks only) ----
    if (k0 + 31 > q0) {
#pragma unroll
      for (int i = 0; i < 8; ++i) {
        const int key0 = k0 + i + hiL * 8;       // key of p0[i]
        if (key0 > q)      p0[i] = -1e30f;
        if (key0 + 16 > q) p1[i] = -1e30f;
      }
    }

    // ---- online softmax (in-lane + one cross-pair shuffle) ----
    float mloc = -1e30f;
#pragma unroll
    for (int i = 0; i < 8; ++i) mloc = fmaxf(mloc, fmaxf(p0[i], p1[i]));
    mloc = fmaxf(mloc, __shfl_xor(mloc, 16, 32));
    const float mnew = fmaxf(m, mloc);
    const float corr = __expf(m - mnew);
    m = mnew;
    l *= corr;
#pragma unroll
    for (int dt = 0; dt < 8; ++dt)
#pragma unroll
      for (int i = 0; i < 8; ++i) acc[dt][i] *= corr;

    float rsum = 0.0f;
#pragma unroll
    for (int i = 0; i < 8; ++i) {
      p0[i] = __expf(p0[i] - m);
      p1[i] = __expf(p1[i] - m);
      rsum += p0[i] + p1[i];
    }
    l += rsum;   // partial (own 16 keys); combined with lane^16 at the end

    // ---- build P^T B-fragment (32 keys x 16 queries) ----
    float o0[8], o1[8];
#pragma unroll
    for (int i = 0; i < 8; ++i) {
      o0[i] = __shfl_xor(p0[i], 16, 32);
      o1[i] = __shfl_xor(p1[i], 16, 32);
    }
    v16bf pb;
#pragma unroll
    for (int e = 0; e < 8; ++e) {
      const float lo = hiL ? o1[e] : p0[e];   // keys hi*16 + e
      const float hs = hiL ? p1[e] : o0[e];   // keys hi*16 + 8 + e
      pb[e]     = (__bf16)lo;
      pb[8 + e] = (__bf16)hs;
    }

    // ---- O^T += V^T(chunk) * P^T ----
#pragma unroll
    for (int dt = 0; dt < 8; ++dt) {
      v16bf va = load_a16(Vh + (size_t)(dt * 16) * SEQ + k0, SEQ);
      acc[dt] = wmma_bf16(va, pb, acc[dt]);
    }
  }

  const float lt   = l + __shfl_xor(l, 16, 32);
  const float rinv = 1.0f / lt;
  const int b0 = bh / NH, h = bh % NH;
  const size_t obase = ((size_t)(b0 * SEQ + q)) * HID + (size_t)h * HD;
#pragma unroll
  for (int dt = 0; dt < 8; ++dt)
#pragma unroll
    for (int i = 0; i < 8; ++i) {
      const int dim = dt * 16 + i + hiL * 8;
      O[obase + dim] = (__bf16)(acc[dt][i] * rinv);
    }
}

// ---------------------------------------------------------------------------
// Output projection: out[T,HID] f32 = Obf16 @ Wp^T.
// One wave computes a 32x128 tile (two row groups share each B fragment).
__global__ void proj_kernel(const __bf16* __restrict__ A,  // T x HID
                            const __bf16* __restrict__ W,  // HID x HID
                            float* __restrict__ out) {
  const int wave = (blockIdx.x * blockDim.x + threadIdx.x) >> 5;
  const int rt = wave >> 4;        // 16 column blocks of 128
  const int cb = wave & 15;
  const int ncol0 = cb * 128;
  const int lane = threadIdx.x & 31;
  const int n = lane & 15, hiL = lane >> 4;

  v8f acc[2][8];
#pragma unroll
  for (int g = 0; g < 2; ++g)
#pragma unroll
    for (int i = 0; i < 8; ++i) acc[g][i] = vzero8();

  const __bf16* Arow0 = A + (size_t)rt * 32 * HID;
  const __bf16* Arow1 = Arow0 + (size_t)16 * HID;
  for (int kk = 0; kk < HID; kk += 32) {
    v16bf a0 = load_a16(Arow0 + kk, HID);
    v16bf a1 = load_a16(Arow1 + kk, HID);
#pragma unroll
    for (int nt = 0; nt < 8; ++nt) {
      v16bf b = load_bT16(W + (size_t)(ncol0 + nt * 16) * HID + kk, HID);
      acc[0][nt] = wmma_bf16(a0, b, acc[0][nt]);
      acc[1][nt] = wmma_bf16(a1, b, acc[1][nt]);
    }
  }
#pragma unroll
  for (int g = 0; g < 2; ++g)
#pragma unroll
    for (int nt = 0; nt < 8; ++nt)
#pragma unroll
      for (int i = 0; i < 8; ++i) {
        const int t   = rt * 32 + g * 16 + i + hiL * 8;
        const int col = ncol0 + nt * 16 + n;
        out[(size_t)t * HID + col] = acc[g][nt][i];
      }
}

// ---------------------------------------------------------------------------
extern "C" void kernel_launch(void* const* d_in, const int* in_sizes, int n_in,
                              void* d_out, int out_size, void* d_ws, size_t ws_size,
                              hipStream_t stream) {
  const float* hidden = (const float*)d_in[0];
  const float* cosb   = (const float*)d_in[1];
  const float* sinb   = (const float*)d_in[2];
  const float* w_attn = (const float*)d_in[3];
  const float* b_attn = (const float*)d_in[4];
  const float* w_proj = (const float*)d_in[5];
  // d_in[6] = cu_seqlens: uniform S, folded into constants.

  __bf16* ws = (__bf16*)d_ws;
  size_t off = 0;
  __bf16* hbf  = ws + off; off += (size_t)T_TOK * HID;
  __bf16* wabf = ws + off; off += (size_t)HID3 * HID;
  __bf16* wpbf = ws + off; off += (size_t)HID * HID;
  __bf16* qws  = ws + off; off += (size_t)NB * NH * SEQ * HD;
  __bf16* kws  = ws + off; off += (size_t)NB * NH * SEQ * HD;
  __bf16* vtws = ws + off; off += (size_t)NB * NH * HD * SEQ;
  __bf16* obf  = ws + off; off += (size_t)T_TOK * HID;

  const int nh = T_TOK * HID;
  const int na = HID3 * HID;
  const int np = HID * HID;
  f32_to_bf16_kernel<<<(nh + 255) / 256, 256, 0, stream>>>(hidden, hbf, nh);
  f32_to_bf16_kernel<<<(na + 255) / 256, 256, 0, stream>>>(w_attn, wabf, na);
  f32_to_bf16_kernel<<<(np + 255) / 256, 256, 0, stream>>>(w_proj, wpbf, np);

  // (T/32)*48 waves, 8 waves/block
  qkv_rope_kernel<<<(T_TOK / 32) * 48 / 8, 256, 0, stream>>>(
      hbf, wabf, b_attn, cosb, sinb, qws, kws, vtws);

  // B*H blocks * (S/128) q-blocks, 8 waves of 16 queries each
  attn_kernel<<<NB * NH * (SEQ / 128), 256, 0, stream>>>(qws, kws, vtws, obf);

  // (T/32)*(HID/128) waves, 8 waves/block
  proj_kernel<<<(T_TOK / 32) * (HID / 128) / 8, 256, 0, stream>>>(
      obf, wpbf, (float*)d_out);
}
